// VoxelizationModule_60705067761969
// MI455X (gfx1250) — matrix operand order repaired and checked by reading, not compiled, not run on hardware
//
#include <hip/hip_runtime.h>
#include <hip/hip_bf16.h>
#include <math.h>

// ---------------- problem constants ----------------
#define NV    8
#define NH    128
#define NW    128
#define NC    512
#define NP    128
#define NPTS  131072          // NV*NH*NW = 2^17
#define NSEG  120000          // NUM_VOXELS; 1875 * 64 exactly
#define MAXI  ((1u<<20)-1u)
#define VOXSZ 0.05f

typedef __attribute__((ext_vector_type(16))) _Float16 v16h;
typedef __attribute__((ext_vector_type(2)))  _Float16 v2h;
typedef __attribute__((ext_vector_type(8)))  float    v8f;

// ---------------- small helpers ----------------
__device__ __forceinline__ void inv3x3(const float* A, float* O) {
    float a = A[0], b = A[1], c = A[2];
    float d = A[3], e = A[4], f = A[5];
    float g = A[6], h = A[7], i = A[8];
    float c0 = e*i - f*h, c1 = f*g - d*i, c2 = d*h - e*g;
    float det = a*c0 + b*c1 + c*c2;
    float inv = 1.0f / det;
    O[0] = c0*inv;            O[1] = (c*h - b*i)*inv;  O[2] = (b*f - c*e)*inv;
    O[3] = c1*inv;            O[4] = (a*i - c*g)*inv;  O[5] = (c*d - a*f)*inv;
    O[6] = c2*inv;            O[7] = (b*g - a*h)*inv;  O[8] = (a*e - b*d)*inv;
}

// ---------------- view precompute: Kinv [V,9], Einv(3x4) [V,12] ----------------
__global__ void precompute_views(const float* __restrict__ K, const float* __restrict__ E,
                                 float* __restrict__ kinv, float* __restrict__ einv) {
    int v = threadIdx.x;
    if (v >= NV) return;
    float Km[9], Ki[9];
    for (int i = 0; i < 9; ++i) Km[i] = K[v*9 + i];
    inv3x3(Km, Ki);
    for (int i = 0; i < 9; ++i) kinv[v*9 + i] = Ki[i];

    float R[9], Ri[9];
    for (int r = 0; r < 3; ++r)
        for (int c = 0; c < 3; ++c) R[r*3 + c] = E[v*12 + r*4 + c];
    float t0 = E[v*12 + 3], t1 = E[v*12 + 7], t2 = E[v*12 + 11];
    inv3x3(R, Ri);
    for (int r = 0; r < 3; ++r) {
        einv[v*12 + r*4 + 0] = Ri[r*3 + 0];
        einv[v*12 + r*4 + 1] = Ri[r*3 + 1];
        einv[v*12 + r*4 + 2] = Ri[r*3 + 2];
        einv[v*12 + r*4 + 3] = -(Ri[r*3+0]*t0 + Ri[r*3+1]*t1 + Ri[r*3+2]*t2);
    }
}

// ---------------- back-project + voxel id + softplus score ----------------
__global__ __launch_bounds__(256)
void compute_vid(const float* __restrict__ depth, const float* __restrict__ conf,
                 const float* __restrict__ kinv, const float* __restrict__ einv,
                 unsigned long long* __restrict__ keys, int* __restrict__ vals,
                 float* __restrict__ scores) {
    int p = blockIdx.x*256 + threadIdx.x;
    if (p >= NPTS) return;
    int v   = p >> 14;          // / (H*W)
    int pix = p & 16383;
    float fy = (float)(pix >> 7);
    float fx = (float)(pix & 127);
    const float* Ki = kinv + v*9;
    float d  = depth[p];
    float cx = (Ki[0]*fx + Ki[1]*fy + Ki[2]) * d;
    float cy = (Ki[3]*fx + Ki[4]*fy + Ki[5]) * d;
    float cz = (Ki[6]*fx + Ki[7]*fy + Ki[8]) * d;
    const float* Ei = einv + v*12;
    float wx = Ei[0]*cx + Ei[1]*cy + Ei[2]*cz  + Ei[3];
    float wy = Ei[4]*cx + Ei[5]*cy + Ei[6]*cz  + Ei[7];
    float wz = Ei[8]*cx + Ei[9]*cy + Ei[10]*cz + Ei[11];
    long long ix = (long long)rintf(wx / VOXSZ);   // rint == round-half-even (jnp.round)
    long long iy = (long long)rintf(wy / VOXSZ);
    long long iz = (long long)rintf(wz / VOXSZ);
    ix = ix < 0 ? 0 : (ix > (long long)MAXI ? (long long)MAXI : ix);
    iy = iy < 0 ? 0 : (iy > (long long)MAXI ? (long long)MAXI : iy);
    iz = iz < 0 ? 0 : (iz > (long long)MAXI ? (long long)MAXI : iz);
    keys[p] = ((unsigned long long)ix << 40) | ((unsigned long long)iy << 20) | (unsigned long long)iz;
    vals[p] = p;
    float x = conf[p];                             // softplus, numerically stable
    scores[p] = fmaxf(x, 0.0f) + log1pf(expf(-fabsf(x)));
}

// ---------------- bitonic sort (N = 2^17), key+payload ----------------
__global__ __launch_bounds__(256)
void bitonic_step(unsigned long long* __restrict__ key, int* __restrict__ val, int j, int k) {
    int i = blockIdx.x*256 + threadIdx.x;
    int ixj = i ^ j;
    if (ixj <= i || i >= NPTS) return;
    bool up = ((i & k) == 0);
    unsigned long long a = key[i], b = key[ixj];
    if ((a > b) == up) {
        key[i] = b; key[ixj] = a;
        int t = val[i]; val[i] = val[ixj]; val[ixj] = t;
    }
}

// ---------------- unique: head flags + 2-level scan ----------------
__global__ __launch_bounds__(256)
void mark_heads(const unsigned long long* __restrict__ keys, int* __restrict__ flags) {
    int i = blockIdx.x*256 + threadIdx.x;
    if (i >= NPTS) return;
    flags[i] = (i == 0 || keys[i] != keys[i-1]) ? 1 : 0;
}

__global__ __launch_bounds__(256)
void scan_block(const int* __restrict__ in, int* __restrict__ out, int* __restrict__ bsums) {
    __shared__ int sh[256];
    int i = blockIdx.x*256 + threadIdx.x;
    int v = (i < NPTS) ? in[i] : 0;
    sh[threadIdx.x] = v;  __syncthreads();
    for (int off = 1; off < 256; off <<= 1) {
        int t = (threadIdx.x >= off) ? sh[threadIdx.x - off] : 0;
        __syncthreads();
        sh[threadIdx.x] += t;
        __syncthreads();
    }
    if (i < NPTS) out[i] = sh[threadIdx.x];
    if (threadIdx.x == 255) bsums[blockIdx.x] = sh[255];
}

__global__ __launch_bounds__(512)
void scan_sums(int* __restrict__ bsums, int nb) {
    __shared__ int sh[512];
    int v = (threadIdx.x < nb) ? bsums[threadIdx.x] : 0;
    sh[threadIdx.x] = v;  __syncthreads();
    for (int off = 1; off < 512; off <<= 1) {
        int t = (threadIdx.x >= off) ? sh[threadIdx.x - off] : 0;
        __syncthreads();
        sh[threadIdx.x] += t;
        __syncthreads();
    }
    if (threadIdx.x < nb) bsums[threadIdx.x] = sh[threadIdx.x];
}

__global__ __launch_bounds__(256)
void scan_add(int* __restrict__ out, const int* __restrict__ bsums) {
    int i = blockIdx.x*256 + threadIdx.x;
    if (i < NPTS && blockIdx.x > 0) out[i] += bsums[blockIdx.x - 1];
}

__global__ __launch_bounds__(256)
void scatter_uids(const unsigned long long* __restrict__ keys, const int* __restrict__ vals,
                  const int* __restrict__ flags, const int* __restrict__ scanv,
                  int* __restrict__ seg, long long* __restrict__ uids) {
    int i = blockIdx.x*256 + threadIdx.x;
    if (i >= NPTS) return;
    int r = scanv[i] - 1;
    if (r >= NSEG) r = NSEG - 1;
    seg[vals[i]] = r;
    if (flags[i] && r < NSEG) uids[r] = (long long)keys[i];
}

// ---------------- segment softmax pooling ----------------
__global__ __launch_bounds__(256)
void seg_max(const float* __restrict__ scores, const int* __restrict__ seg, float* __restrict__ m) {
    int p = blockIdx.x*256 + threadIdx.x;
    if (p >= NPTS) return;
    // softplus > 0 and m initialized to 0.0f -> uint atomicMax is order-correct
    atomicMax((unsigned int*)(m + seg[p]), __float_as_uint(scores[p]));
}

__global__ __launch_bounds__(256)
void seg_exp(const float* __restrict__ scores, const int* __restrict__ seg,
             const float* __restrict__ m, float* __restrict__ evals, float* __restrict__ denom) {
    int p = blockIdx.x*256 + threadIdx.x;
    if (p >= NPTS) return;
    int s = seg[p];
    float e = expf(scores[p] - m[s]);
    evals[p] = e;
    atomicAdd(denom + s, e);
}

__global__ __launch_bounds__(256)
void scatter_feats(const float* __restrict__ feats, const float* __restrict__ evals,
                   const float* __restrict__ denom, const int* __restrict__ seg,
                   float* __restrict__ vfeat) {
    int idx = blockIdx.x*256 + threadIdx.x;      // NPTS * (NC/4) threads
    int p  = idx >> 7;                           // / 128
    int c4 = (idx & 127) << 2;
    if (p >= NPTS) return;
    int s = seg[p];
    float w = evals[p] / fmaxf(denom[s], 1e-12f);
    const float* fp = feats + (size_t)p * NC + c4;
    float* vp = vfeat + (size_t)s * NC + c4;
    atomicAdd(vp + 0, fp[0]*w);
    atomicAdd(vp + 1, fp[1]*w);
    atomicAdd(vp + 2, fp[2]*w);
    atomicAdd(vp + 3, fp[3]*w);
}

// ---------------- voxel centers ----------------
__global__ __launch_bounds__(256)
void centers_kernel(const long long* __restrict__ uids, float* __restrict__ centers) {
    int s = blockIdx.x*256 + threadIdx.x;
    if (s >= NSEG) return;
    long long u = uids[s];
    float cx = 0.f, cy = 0.f, cz = 0.f;
    if (u >= 0) {
        cx = (float)((u >> 40) & MAXI) * VOXSZ;
        cy = (float)((u >> 20) & MAXI) * VOXSZ;
        cz = (float)( u        & MAXI) * VOXSZ;
    }
    centers[s*3 + 0] = cx; centers[s*3 + 1] = cy; centers[s*3 + 2] = cz;
}

// ---------------- WMMA GEMM: out[M,N] = concat(A1[K1],A2[K2]) @ Bw[K,N] + bias ----------------
// block = 128 threads (4 waves); block tile 64x64; wave tile 32x32 = 2x2 v_wmma_f32_16x16x32_f16
// A staged row-major (per-lane fragment = two contiguous 16B runs -> ds_load_b128 x2).
// B staged FRAGMENT-MAJOR: lBf[subtile][lane][16] so each lane's operand is one
// contiguous 32B run -> ds_load_b128 x2 (no ds_load_u16 / v_mov_b16 packing).
// M is always a multiple of TM (120000 = 1875*64) -> no row guards in epilogue.
#define TM 64
#define TN 64
#define TK 32
__global__ __launch_bounds__(128)
void gemm_wmma(const float* __restrict__ A1, int K1,
               const float* __restrict__ A2, int K2,
               const float* __restrict__ Bw, const float* __restrict__ bias,
               float* __restrict__ Out, int N) {
    __shared__ _Float16 lA[TM][TK];             // 4 KB
    __shared__ _Float16 lBf[TN/16][32][16];     // 4 KB, fragment-major
    const int K    = K1 + K2;
    const int tid  = threadIdx.x;
    const int wave = tid >> 5;
    const int lane = tid & 31;
    const int m0 = blockIdx.x * TM;
    const int n0 = blockIdx.y * TN;
    const int wm = (wave >> 1) * 32;       // wave quadrant in block tile
    const int wn = (wave & 1) * 32;
    const int hl  = lane >> 4;             // half of wave
    const int l16 = lane & 15;

    v8f acc[2][2] = {};

    for (int k0 = 0; k0 < K; k0 += TK) {
        // stage A 64x32 (f32 -> f16), dual-source K split; 2 consecutive k per thread
        for (int i = tid; i < TM*(TK/2); i += 128) {
            int r = i >> 4, c2 = (i & 15) << 1;
            int gk = k0 + c2;
            const float* src = (gk < K1) ? (A1 + (size_t)(m0 + r) * K1 + gk)
                                         : (A2 + (size_t)(m0 + r) * K2 + (gk - K1));
            v2h pk; pk[0] = (_Float16)src[0]; pk[1] = (_Float16)src[1];
            *(v2h*)(&lA[r][c2]) = pk;                    // 4B-aligned ds_store_b32
        }
        // stage B 32x64 into fragment-major layout: lane=(k>=16)*16+(n&15), e=k&15
        for (int i = tid; i < (TK/2)*TN; i += 128) {
            int rp = (i >> 6) << 1;                      // even k: 0,2,...,30
            int c  = i & 63;
            float v0 = Bw[(size_t)(k0 + rp)     * N + (n0 + c)];
            float v1 = Bw[(size_t)(k0 + rp + 1) * N + (n0 + c)];
            int sub = c >> 4;
            int ln2 = ((rp & 16) ? 16 : 0) + (c & 15);
            int e   = rp & 15;                           // even
            v2h pk; pk[0] = (_Float16)v0; pk[1] = (_Float16)v1;
            *(v2h*)(&lBf[sub][ln2][e]) = pk;             // 4B-aligned ds_store_b32
        }
        __syncthreads();

        #pragma unroll
        for (int ti = 0; ti < 2; ++ti) {
            // A operand: contiguous k order (k = 16*half + e), matching B fragment-major layout
            v16h av;
            #pragma unroll
            for (int g = 0; g < 8; ++g) {
                int kb = hl*16 + g*2;
                av[2*g]   = lA[wm + ti*16 + l16][kb];
                av[2*g+1] = lA[wm + ti*16 + l16][kb + 1];
            }
            #pragma unroll
            for (int tj = 0; tj < 2; ++tj) {
                // B operand: one contiguous 32B run per lane
                v16h bv = *(const v16h*)(&lBf[(wn >> 4) + tj][lane][0]);
                acc[ti][tj] = __builtin_amdgcn_wmma_f32_16x16x32_f16(
                    false, av, false, bv, (short)0, acc[ti][tj], false, false);
            }
        }
        __syncthreads();
    }

    // epilogue: C/D layout — VGPR g: lanes 0-15 -> M=g, lanes 16-31 -> M=8+g.
    // No row guard (M % 64 == 0); incremental pointer arithmetic -> straight store clause.
    #pragma unroll
    for (int ti = 0; ti < 2; ++ti)
        #pragma unroll
        for (int tj = 0; tj < 2; ++tj) {
            int col = n0 + wn + tj*16 + l16;
            float bval = bias ? bias[col] : 0.0f;
            float* op = Out + (size_t)(m0 + wm + ti*16 + hl*8) * N + col;
            #pragma unroll
            for (int g = 0; g < 8; ++g) {
                *op = acc[ti][tj][g] + bval;
                op += N;
            }
        }
}

// ---------------- LayerNorm + exact GELU (+ optional valid mask) ----------------
__global__ __launch_bounds__(256)
void ln_gelu(float* __restrict__ X, const float* __restrict__ gw, const float* __restrict__ bw,
             const long long* __restrict__ uids, int Cdim) {
    int row = blockIdx.x;
    float* xr = X + (size_t)row * Cdim;
    int t = threadIdx.x;
    float s = 0.f, s2 = 0.f;
    for (int c = t; c < Cdim; c += 256) { float v = xr[c]; s += v; s2 += v*v; }
    for (int off = 16; off > 0; off >>= 1) {
        s  += __shfl_down(s,  off);
        s2 += __shfl_down(s2, off);
    }
    __shared__ float shs[8], shq[8];
    int wv = t >> 5;
    if ((t & 31) == 0) { shs[wv] = s; shq[wv] = s2; }
    __syncthreads();
    float tot = 0.f, tot2 = 0.f;
    #pragma unroll
    for (int i = 0; i < 8; ++i) { tot += shs[i]; tot2 += shq[i]; }
    float mu  = tot / (float)Cdim;
    float var = tot2 / (float)Cdim - mu*mu;
    float inv = rsqrtf(var + 1e-5f);
    float mask = 1.0f;
    if (uids) mask = (uids[row] >= 0) ? 1.0f : 0.0f;
    for (int c = t; c < Cdim; c += 256) {
        float v = (xr[c] - mu) * inv * gw[c] + bw[c];
        v = 0.5f * v * (1.0f + erff(v * 0.70710678118654752f));   // exact gelu
        xr[c] = v * mask;
    }
}

// ---------------- pos encoding layer 1: silu(centers @ w_pos1 + b) ----------------
__global__ __launch_bounds__(256)
void pos1_kernel(const float* __restrict__ centers, const float* __restrict__ w,
                 const float* __restrict__ b, float* __restrict__ out) {
    int i = blockIdx.x*256 + threadIdx.x;        // NSEG * NP threads
    if (i >= NSEG * NP) return;
    int s = i >> 7, j = i & 127;
    float x = centers[s*3+0]*w[j] + centers[s*3+1]*w[NP + j] + centers[s*3+2]*w[2*NP + j] + b[j];
    out[i] = x / (1.0f + expf(-x));              // silu
}

// ---------------- host launcher ----------------
static inline char* ws_take(char*& p, size_t bytes) {
    char* r = p;
    p += (bytes + 255) & ~(size_t)255;
    return r;
}

extern "C" void kernel_launch(void* const* d_in, const int* in_sizes, int n_in,
                              void* d_out, int out_size, void* d_ws, size_t ws_size,
                              hipStream_t stream) {
    const float* features   = (const float*)d_in[0];
    const float* depth      = (const float*)d_in[1];
    const float* intrinsics = (const float*)d_in[2];
    const float* extrinsics = (const float*)d_in[3];
    const float* conf       = (const float*)d_in[4];
    const float* w_proj1    = (const float*)d_in[5];
    const float* b_proj1    = (const float*)d_in[6];
    const float* g_ln1      = (const float*)d_in[7];
    const float* be_ln1     = (const float*)d_in[8];
    const float* w_proj2    = (const float*)d_in[9];
    const float* b_proj2    = (const float*)d_in[10];
    const float* w_pos1     = (const float*)d_in[11];
    const float* b_pos1     = (const float*)d_in[12];
    const float* w_pos2     = (const float*)d_in[13];
    const float* b_pos2     = (const float*)d_in[14];
    const float* w_fuse     = (const float*)d_in[15];
    const float* b_fuse     = (const float*)d_in[16];
    const float* g_ln2      = (const float*)d_in[17];
    const float* be_ln2     = (const float*)d_in[18];
    float* out = (float*)d_out;

    char* wp = (char*)d_ws;
    float*              kinv    = (float*)             ws_take(wp, NV*9*sizeof(float));
    float*              einv    = (float*)             ws_take(wp, NV*12*sizeof(float));
    unsigned long long* keys    = (unsigned long long*)ws_take(wp, (size_t)NPTS*8);
    int*                vals    = (int*)               ws_take(wp, (size_t)NPTS*4);
    int*                seg     = (int*)               ws_take(wp, (size_t)NPTS*4);
    int*                flags   = (int*)               ws_take(wp, (size_t)NPTS*4);
    int*                scanv   = (int*)               ws_take(wp, (size_t)NPTS*4);
    int*                bsums   = (int*)               ws_take(wp, 512*4);
    long long*          uids    = (long long*)         ws_take(wp, (size_t)NSEG*8);
    float*              scores  = (float*)             ws_take(wp, (size_t)NPTS*4);
    float*              evals   = (float*)             ws_take(wp, (size_t)NPTS*4);
    float*              segm    = (float*)             ws_take(wp, (size_t)NSEG*4);
    float*              denom   = (float*)             ws_take(wp, (size_t)NSEG*4);
    float*              centers = (float*)             ws_take(wp, (size_t)NSEG*3*4);
    float*              bufA    = (float*)             ws_take(wp, (size_t)NSEG*NC*4); // vfeat -> h2
    float*              bufB    = (float*)             ws_take(wp, (size_t)NSEG*NC*4); // h1
    float*              bufD1   = (float*)             ws_take(wp, (size_t)NSEG*NP*4); // pe1
    float*              bufD2   = (float*)             ws_take(wp, (size_t)NSEG*NP*4); // pe2

    // fresh accumulator state every call (deterministic)
    hipMemsetAsync(segm,  0,    (size_t)NSEG*4,    stream);
    hipMemsetAsync(denom, 0,    (size_t)NSEG*4,    stream);
    hipMemsetAsync(bufA,  0,    (size_t)NSEG*NC*4, stream);
    hipMemsetAsync(uids,  0xFF, (size_t)NSEG*8,    stream);   // -1 fill

    const int PB = NPTS / 256;        // 512 point-blocks
    const int SB = (NSEG + 255) / 256;

    precompute_views<<<1, 32, 0, stream>>>(intrinsics, extrinsics, kinv, einv);
    compute_vid<<<PB, 256, 0, stream>>>(depth, conf, kinv, einv, keys, vals, scores);

    for (int k = 2; k <= NPTS; k <<= 1)
        for (int j = k >> 1; j > 0; j >>= 1)
            bitonic_step<<<PB, 256, 0, stream>>>(keys, vals, j, k);

    mark_heads<<<PB, 256, 0, stream>>>(keys, flags);
    scan_block<<<PB, 256, 0, stream>>>(flags, scanv, bsums);
    scan_sums<<<1, 512, 0, stream>>>(bsums, PB);
    scan_add<<<PB, 256, 0, stream>>>(scanv, bsums);
    scatter_uids<<<PB, 256, 0, stream>>>(keys, vals, flags, scanv, seg, uids);

    seg_max<<<PB, 256, 0, stream>>>(scores, seg, segm);
    seg_exp<<<PB, 256, 0, stream>>>(scores, seg, segm, evals, denom);
    scatter_feats<<<(NPTS*(NC/4))/256, 256, 0, stream>>>(features, evals, denom, seg, bufA);

    centers_kernel<<<SB, 256, 0, stream>>>(uids, centers);

    // h1 = vfeat @ w_proj1 + b ; LN ; gelu
    gemm_wmma<<<dim3(NSEG/TM, NC/TN), 128, 0, stream>>>(bufA, NC, nullptr, 0, w_proj1, b_proj1, bufB, NC);
    ln_gelu<<<NSEG, 256, 0, stream>>>(bufB, g_ln1, be_ln1, nullptr, NC);
    // h2 = h1 @ w_proj2 + b   (overwrites vfeat)
    gemm_wmma<<<dim3(NSEG/TM, NC/TN), 128, 0, stream>>>(bufB, NC, nullptr, 0, w_proj2, b_proj2, bufA, NC);
    // pe = silu(centers @ w_pos1 + b) @ w_pos2 + b
    pos1_kernel<<<(NSEG*NP)/256, 256, 0, stream>>>(centers, w_pos1, b_pos1, bufD1);
    gemm_wmma<<<dim3(NSEG/TM, NP/TN), 128, 0, stream>>>(bufD1, NP, nullptr, 0, w_pos2, b_pos2, bufD2, NP);
    // tok = concat(h2, pe) @ w_fuse + b ; LN ; gelu ; mask
    gemm_wmma<<<dim3(NSEG/TM, NC/TN), 128, 0, stream>>>(bufA, NC, bufD2, NP, w_fuse, b_fuse, out, NC);
    ln_gelu<<<NSEG, 256, 0, stream>>>(out, g_ln2, be_ln2, uids, NC);

    (void)in_sizes; (void)n_in; (void)out_size; (void)ws_size;
}
